// decoder0_pacpool_42872363549301
// MI455X (gfx1250) — matrix operand order repaired and checked by reading, not compiled, not run on hardware
//
#include <hip/hip_runtime.h>

// CDNA5 / gfx1250: wave32, WMMA (no MFMA).
typedef __attribute__((ext_vector_type(16))) __bf16 v16bf;
typedef __attribute__((ext_vector_type(8)))  __bf16 v8bf;
typedef __attribute__((ext_vector_type(8)))  float  v8f;

// ---------------------------------------------------------------------------
// Repack conv weights [oc][ic][3][3] (fp32, tap fastest) into two bf16 planes
// [tap][oc][ic] (hi + lo of the fp32 split).  Done once per layer, so the
// conv kernel's B operand is fragment-ready in global memory: a lane's
// 16-element B fragment is 16 contiguous bf16 (two global_load_b128),
// no LDS staging and no per-block re-conversion.
// ---------------------------------------------------------------------------
__global__ void weight_repack(const float* __restrict__ w, __bf16* __restrict__ wh,
                              __bf16* __restrict__ wl, int Cout, int Cin)
{
  long i = (long)blockIdx.x * blockDim.x + threadIdx.x;
  long total = (long)Cout * Cin * 9;
  if (i >= total) return;
  int tap = (int)(i % 9); long r = i / 9;
  int ic = (int)(r % Cin); int oc = (int)(r / Cin);
  float v = w[i];
  __bf16 h = (__bf16)v;
  long o = ((long)tap * Cout + oc) * Cin + ic;
  wh[o] = h;
  wl[o] = (__bf16)(v - (float)h);
}

// ---------------------------------------------------------------------------
// Implicit-GEMM 3x3 conv, pad=1, fp32 in / fp32 out, bf16 hi/lo split on the
// v_wmma_f32_16x16x32_bf16 pipe (f32 accumulate).
//   M = B*H*W (multiple of 64 for every layer)
//   N = Cout  (multiple of 64)
//   K = Cin*9, iterated TAP-MAJOR: 9 taps x (Cin/32) chunks of 32 channels.
// Block: 256 threads = 8 wave32, 64x64 output tile, wave = 16x32 sub-tile.
//
// A (activations) is staged through LDS in fragment order:
//  - rows stored in natural K order, so the per-lane fragment
//    (K = 0..15 or 16..31) is elements [0,16) / [16,32).
//  - each thread owns ONE pixel (ml = t&63) and 8 *contiguous* k
//    (kb = t>>6, k in [8kb, 8kb+8)); stored unpermuted as one aligned
//    8-block, so hi/lo are each one ds_store_b128.
//  - zero-padding handled with unpredicated loads from a safe base +
//    v_cndmask (no exec toggling), running pointer (no per-chunk i64 mul).
// B (weights) is read directly from the repacked bf16 planes in global
// memory (L2-resident: <=19 MB per layer vs 192 MB L2), two
// global_load_b128 per fragment.
// ---------------------------------------------------------------------------
__global__ __launch_bounds__(256) void conv3x3_wmma(
    const float* __restrict__ in, const __bf16* __restrict__ wh,
    const __bf16* __restrict__ wl, const float* __restrict__ bias,
    float* __restrict__ out, int B, int Cin, int H, int W, int Cout)
{
  (void)B;
  const int HW = H * W;
  const int M0 = blockIdx.x * 64;
  const int N0 = blockIdx.y * 64;

  __shared__ __align__(16) __bf16 As[2][64][40];  // [hi/lo][pixel][k]

  const int t    = threadIdx.x;
  const int lane = t & 31;
  const int wm   = (t >> 5) & 3;     // wave row (0..3) -> 16 pixels each
  const int wn   = t >> 7;           // wave col (0..1) -> 32 ocs each
  const int half = (lane < 16) ? 0 : 16;

  // ---- A staging: one pixel per thread, 8 contiguous k starting at 8*kb
  const int ml  = t & 63;
  const int kb  = t >> 6;                       // 0..3
  const int m   = M0 + ml;
  const int pb  = m / HW;
  const int prm = m - pb * HW;
  const int py  = prm / W;
  const int px  = prm - py * W;
  const float* __restrict__ inb = in + (long)(pb * Cin) * HW + (long)(kb * 8) * HW;
  const int pb8 = kb * 8;   // K blocks stored in natural (contiguous) order
  const long chunkStep = (long)32 * HW;

  // ---- B fragment row offsets (hoisted; only ic0 varies per chunk)
  const long rowoff0 = (long)(wn * 32 + (lane & 15)) * Cin + half;
  const long rowoff1 = rowoff0 + (long)16 * Cin;

  v8f acc[2] = {};

  int ky = 0, kx = 0;
  for (int tap = 0; tap < 9; ++tap) {
    const int iy = py + ky - 1, ix = px + kx - 1;
    const bool valid = (unsigned)iy < (unsigned)H && (unsigned)ix < (unsigned)W;
    // Safe base: when out of bounds, read (and discard) from the pixel's own
    // channel plane instead — always in-bounds for every (ic0+l)*HW offset.
    const float* __restrict__ pc = valid ? (inb + (iy * W + ix)) : inb;
    const __bf16* __restrict__ pWh = wh + ((long)tap * Cout + N0) * Cin;
    const __bf16* __restrict__ pWl = wl + ((long)tap * Cout + N0) * Cin;

    for (int ic0 = 0; ic0 < Cin; ic0 += 32) {
      // ---- stage A: 8 unpredicated coalesced loads + select, pack,
      //      2x ds_store_b128
      float v[8];
#pragma unroll
      for (int l = 0; l < 8; ++l) {
        float raw = pc[(long)l * HW];        // l*HW is loop-invariant (LICM)
        v[l] = valid ? raw : 0.f;
      }
      v8bf hv, lv;
#pragma unroll
      for (int l = 0; l < 8; ++l) {
        __bf16 h = (__bf16)v[l];
        hv[l] = h;
        lv[l] = (__bf16)(v[l] - (float)h);
      }
      *(v8bf*)&As[0][ml][pb8] = hv;
      *(v8bf*)&As[1][ml][pb8] = lv;
      __syncthreads();

      // ---- A fragments: two contiguous 16B LDS loads each
      const int arow = wm * 16 + (lane & 15);
      union { v16bf v; v8bf h[2]; } ua, ub;
      ua.h[0] = *(const v8bf*)&As[0][arow][half];
      ua.h[1] = *(const v8bf*)&As[0][arow][half + 8];
      v16bf a_h = ua.v;
      ua.h[0] = *(const v8bf*)&As[1][arow][half];
      ua.h[1] = *(const v8bf*)&As[1][arow][half + 8];
      v16bf a_l = ua.v;

      // ---- two 16x16 N sub-tiles; B fragments straight from global (L2)
#pragma unroll
      for (int s = 0; s < 2; ++s) {
        const long rb = (s == 0 ? rowoff0 : rowoff1) + ic0;
        ub.h[0] = *(const v8bf*)(pWh + rb);
        ub.h[1] = *(const v8bf*)(pWh + rb + 8);
        v16bf b_h = ub.v;
        ub.h[0] = *(const v8bf*)(pWl + rb);
        ub.h[1] = *(const v8bf*)(pWl + rb + 8);
        v16bf b_l = ub.v;
        // split product: hi*hi + hi*lo + lo*hi  (lo*lo below f32 rounding)
        acc[s] = __builtin_amdgcn_wmma_f32_16x16x32_bf16(false, a_h, false, b_h,
                                                         (short)0, acc[s], false, false);
        acc[s] = __builtin_amdgcn_wmma_f32_16x16x32_bf16(false, a_h, false, b_l,
                                                         (short)0, acc[s], false, false);
        acc[s] = __builtin_amdgcn_wmma_f32_16x16x32_bf16(false, a_l, false, b_h,
                                                         (short)0, acc[s], false, false);
      }
      __syncthreads();
      pc += chunkStep;
    }
    ++kx; if (kx == 3) { kx = 0; ++ky; }
  }

  // ---- epilogue: C/D layout (VGPR j: M=j / j+8 by lane half, N=lane%16)
#pragma unroll
  for (int s = 0; s < 2; ++s) {
    int n = N0 + wn * 32 + s * 16 + (lane & 15);
    float bv = bias[n];
#pragma unroll
    for (int j = 0; j < 8; ++j) {
      int mo = M0 + wm * 16 + j + ((lane < 16) ? 0 : 8);
      int bo = mo / HW;  int rm = mo - bo * HW;
      out[(long)(bo * Cout + n) * HW + rm] = acc[s][j] + bv;
    }
  }
}

// ---------------------------------------------------------------------------
// GroupNorm + ReLU, fused; writes into a concat buffer (Cdst channels) at
// channel offset 0.  One block per (batch, group).
// ---------------------------------------------------------------------------
__global__ __launch_bounds__(1024) void gn_relu(
    const float* __restrict__ src, const float* __restrict__ gamma,
    const float* __restrict__ beta, float* __restrict__ dst,
    int C, int Gn, int HW, int Cdst)
{
  const int b = blockIdx.x, g = blockIdx.y;
  const int cpg = C / Gn;
  const long base = (long)(b * C + g * cpg) * HW;
  const long n = (long)cpg * HW;
  float s = 0.f, ss = 0.f;
  for (long i = threadIdx.x; i < n; i += 1024) {
    float v = src[base + i];
    s += v; ss += v * v;
  }
  __shared__ float rs[1024], rq[1024];
  rs[threadIdx.x] = s; rq[threadIdx.x] = ss;
  __syncthreads();
  for (int o = 512; o > 0; o >>= 1) {
    if ((int)threadIdx.x < o) {
      rs[threadIdx.x] += rs[threadIdx.x + o];
      rq[threadIdx.x] += rq[threadIdx.x + o];
    }
    __syncthreads();
  }
  const float mean = rs[0] / (float)n;
  const float var  = rq[0] / (float)n - mean * mean;
  const float rstd = rsqrtf(var + 1e-5f);
  for (int c0 = 0; c0 < cpg; ++c0) {
    const int c = g * cpg + c0;
    const float sc = rstd * gamma[c];
    const float sb = beta[c] - mean * sc;
    const float* __restrict__ sp = src + base + (long)c0 * HW;
    float* __restrict__ dp = dst + (long)(b * Cdst + c) * HW;
    for (int i = threadIdx.x; i < HW; i += 1024)
      dp[i] = fmaxf(sp[i] * sc + sb, 0.f);
  }
}

// ---------------------------------------------------------------------------
// Copy (B,Cs,HW) tensor into concat buffer (B,Cdst,HW) at channel offset.
// ---------------------------------------------------------------------------
__global__ void concat_copy(const float* __restrict__ src, float* __restrict__ dst,
                            int B, int Cs, int HW, int Cdst, int coff)
{
  (void)B;
  long i = (long)blockIdx.x * blockDim.x + threadIdx.x;
  long total = (long)B * Cs * HW;
  if (i >= total) return;
  int p = (int)(i % HW); long r = i / HW;
  int c = (int)(r % Cs); int b = (int)(r / Cs);
  dst[(long)(b * Cdst + coff + c) * HW + p] = src[i];
}

// ---------------------------------------------------------------------------
// jax.image.resize 'linear' (antialias triangle, half-pixel centers,
// weight renormalization at the edges).  Exact bilinear for 2x upsample.
// ---------------------------------------------------------------------------
__global__ void resize_linear(const float* __restrict__ src, float* __restrict__ dst,
                              int B, int C, int Hi, int Wi, int Ho, int Wo)
{
  (void)B;
  long i = (long)blockIdx.x * blockDim.x + threadIdx.x;
  long total = (long)B * C * Ho * Wo;
  if (i >= total) return;
  int wo = (int)(i % Wo); long r = i / Wo;
  int ho = (int)(r % Ho); r /= Ho;
  int c  = (int)(r % C);  int b = (int)(r / C);
  float sy = (float)Hi / (float)Ho, sx = (float)Wi / (float)Wo;
  float cy = (ho + 0.5f) * sy - 0.5f;
  float cx = (wo + 0.5f) * sx - 0.5f;
  float ry = fmaxf(sy, 1.f), rx = fmaxf(sx, 1.f);
  int y0 = max(0, (int)ceilf(cy - ry)), y1 = min(Hi - 1, (int)floorf(cy + ry));
  int x0 = max(0, (int)ceilf(cx - rx)), x1 = min(Wi - 1, (int)floorf(cx + rx));
  const float* p = src + (long)(b * C + c) * Hi * Wi;
  float acc = 0.f, wsum = 0.f;
  for (int y = y0; y <= y1; ++y) {
    float wy = 1.f - fabsf((float)y - cy) / ry;
    if (wy <= 0.f) continue;
    for (int x = x0; x <= x1; ++x) {
      float wx = 1.f - fabsf((float)x - cx) / rx;
      if (wx <= 0.f) continue;
      float w = wy * wx;
      acc += w * p[y * Wi + x];
      wsum += w;
    }
  }
  dst[i] = acc / wsum;
}

// ---------------------------------------------------------------------------
// g (B,4,H,W) already holds linearly-resized emb; multiply by (2*mask-1)
// with nearest-resized mask (Hm,Wm source).
// ---------------------------------------------------------------------------
__global__ void guide_mask(float* __restrict__ g, const float* __restrict__ mask,
                           int B, int H, int W, int Hm, int Wm)
{
  (void)B;
  long i = (long)blockIdx.x * blockDim.x + threadIdx.x;
  long total = (long)B * 4 * H * W;
  if (i >= total) return;
  int w = (int)(i % W); long r = i / W;
  int h = (int)(r % H); r /= H;
  int b = (int)(r / 4);
  int my = min(Hm - 1, (int)(((h + 0.5f) * (float)Hm) / (float)H));
  int mx = min(Wm - 1, (int)(((w + 0.5f) * (float)Wm) / (float)W));
  float m = mask[((long)b * Hm + my) * Wm + mx];
  g[i] *= (2.f * m - 1.f);
}

// ---------------------------------------------------------------------------
// PAC pooling (k=3, zero padding on x and g, per-pixel Gaussian guide
// weights over 4 channels).  Reads channels [0,Cx) of xin, accumulates
// scale * num/(den+1e-8) into channels [Cx,2*Cx).
// ---------------------------------------------------------------------------
__global__ void pac_pool(const float* __restrict__ xin, const float* __restrict__ g,
                         float* __restrict__ dst, int B, int Ctot, int Cx,
                         int H, int W, int dil, float scale, int init)
{
  (void)B;
  long i = (long)blockIdx.x * blockDim.x + threadIdx.x;
  long total = (long)B * H * W;
  if (i >= total) return;
  const int HW = H * W;
  int b = (int)(i / HW); int r = (int)(i % HW);
  int y = r / W, x = r % W;
  const float* gb = g + (long)b * 4 * HW;
  float gc[4];
#pragma unroll
  for (int c = 0; c < 4; ++c) gc[c] = gb[c * HW + r];
  float wt[9]; float den = 0.f;
#pragma unroll
  for (int ti = 0; ti < 3; ++ti)
#pragma unroll
    for (int tj = 0; tj < 3; ++tj) {
      int yy = y + (ti - 1) * dil, xx = x + (tj - 1) * dil;
      float d2 = 0.f;
      if ((unsigned)yy < (unsigned)H && (unsigned)xx < (unsigned)W) {
#pragma unroll
        for (int c = 0; c < 4; ++c) {
          float d = gb[c * HW + yy * W + xx] - gc[c];
          d2 += d * d;
        }
      } else {               // padded guide is 0 => diff = -g_center
#pragma unroll
        for (int c = 0; c < 4; ++c) d2 += gc[c] * gc[c];
      }
      float w = expf(-0.5f * d2);
      wt[ti * 3 + tj] = w;
      den += w;
    }
  const float rden = scale / (den + 1e-8f);
  for (int c = 0; c < Cx; ++c) {
    const float* xc = xin + (long)(b * Ctot + c) * HW;
    float num = 0.f;
#pragma unroll
    for (int ti = 0; ti < 3; ++ti)
#pragma unroll
      for (int tj = 0; tj < 3; ++tj) {
        int yy = y + (ti - 1) * dil, xx = x + (tj - 1) * dil;
        if ((unsigned)yy < (unsigned)H && (unsigned)xx < (unsigned)W)
          num += wt[ti * 3 + tj] * xc[yy * W + xx];
      }
    long o = (long)(b * Ctot + Cx + c) * HW + r;
    float val = num * rden;
    dst[o] = init ? val : (dst[o] + val);
  }
}

// ---------------------------------------------------------------------------
// Final conv: edge-padded 3x3, 64 -> 3 channels (too thin for WMMA).
// ---------------------------------------------------------------------------
__global__ void final_conv(const float* __restrict__ xin, const float* __restrict__ wf,
                           const float* __restrict__ bf, float* __restrict__ out,
                           int B, int H, int W)
{
  (void)B;
  long i = (long)blockIdx.x * blockDim.x + threadIdx.x;
  long total = (long)B * H * W;
  if (i >= total) return;
  const int HW = H * W;
  int b = (int)(i / HW); int r = (int)(i % HW);
  int y = r / W, x = r % W;
  float a0 = bf[0], a1 = bf[1], a2 = bf[2];
  for (int ic = 0; ic < 64; ++ic) {
    const float* p = xin + (long)(b * 64 + ic) * HW;
#pragma unroll
    for (int ky = 0; ky < 3; ++ky) {
      int yy = min(max(y + ky - 1, 0), H - 1);
#pragma unroll
      for (int kx = 0; kx < 3; ++kx) {
        int xx = min(max(x + kx - 1, 0), W - 1);
        float v = p[yy * W + xx];
        int wb = (ic * 3 + ky) * 3 + kx;
        a0 += v * wf[wb];
        a1 += v * wf[576 + wb];
        a2 += v * wf[1152 + wb];
      }
    }
  }
  out[(long)(b * 3 + 0) * HW + r] = a0;
  out[(long)(b * 3 + 1) * HW + r] = a1;
  out[(long)(b * 3 + 2) * HW + r] = a2;
}

// ---------------------------------------------------------------------------
// Orchestration.  Workspace (floats, bump-allocated, reused across stages):
//   XIN  : max concat buffer, 41,943,040   (also holds dx6)
//   U    : max upsample out,  167,772,160
//   T    : max conv out,      41,943,040
//   G    : guide,             655,360
//   WREP : repacked bf16 weights (hi+lo planes), 4,718,592 floats of space
// total ~= 1.03 GB.
// ---------------------------------------------------------------------------
extern "C" void kernel_launch(void* const* d_in, const int* in_sizes, int n_in,
                              void* d_out, int out_size, void* d_ws, size_t ws_size,
                              hipStream_t stream)
{
  (void)in_sizes; (void)n_in; (void)out_size; (void)ws_size;
  const float* x1   = (const float*)d_in[1];
  const float* x2   = (const float*)d_in[2];
  const float* x3   = (const float*)d_in[3];
  const float* x4   = (const float*)d_in[4];
  const float* x5   = (const float*)d_in[5];
  const float* x6   = (const float*)d_in[6];
  const float* emb  = (const float*)d_in[7];
  const float* mask = (const float*)d_in[8];
  const float *Wc[6], *Bc[6], *Ga[6], *Be[6];
  for (int l = 0; l < 6; ++l) {
    Wc[l] = (const float*)d_in[9 + 4 * l];
    Bc[l] = (const float*)d_in[10 + 4 * l];
    Ga[l] = (const float*)d_in[11 + 4 * l];
    Be[l] = (const float*)d_in[12 + 4 * l];
  }
  const float* wf = (const float*)d_in[33];
  const float* bf = (const float*)d_in[34];
  float* out = (float*)d_out;

  float* XIN  = (float*)d_ws;
  float* U    = XIN + 41943040L;
  float* T    = U   + 167772160L;
  float* Gd   = T   + 41943040L;
  float* WREP = Gd  + 655360L;

  const int B = 8;
#define EWG(n) dim3((unsigned)(((n) + 255) / 256)), dim3(256)
#define REPACK(l, CO, CI)                                                        \
  __bf16* WH = (__bf16*)WREP;                                                    \
  __bf16* WL = WH + (long)(CO) * (CI) * 9;                                       \
  weight_repack<<<EWG((long)(CO) * (CI) * 9), 0, stream>>>(Wc[l], WH, WL, CO, CI)

  // ---- Stage 1: dx1 = relu(gn32(conv1(x6))); xin1 = [dx1, x5]; u1 = up2(xin1)
  { REPACK(0, 512, 1024);
    conv3x3_wmma<<<dim3(B * 80 / 64, 512 / 64), 256, 0, stream>>>(x6, WH, WL, Bc[0], T, B, 1024, 8, 10, 512); }
  gn_relu<<<dim3(B, 32), 1024, 0, stream>>>(T, Ga[0], Be[0], XIN, 512, 32, 80, 1024);
  concat_copy<<<EWG((long)B * 512 * 80), 0, stream>>>(x5, XIN, B, 512, 80, 1024, 512);
  resize_linear<<<EWG((long)B * 1024 * 320), 0, stream>>>(XIN, U, B, 1024, 8, 10, 16, 20);

  // ---- Stage 2: dx2; xin2 = [dx2, x4]; u2 = up2(xin2)
  { REPACK(1, 256, 1024);
    conv3x3_wmma<<<dim3(B * 320 / 64, 256 / 64), 256, 0, stream>>>(U, WH, WL, Bc[1], T, B, 1024, 16, 20, 256); }
  gn_relu<<<dim3(B, 16), 1024, 0, stream>>>(T, Ga[1], Be[1], XIN, 256, 16, 320, 512);
  concat_copy<<<EWG((long)B * 256 * 320), 0, stream>>>(x4, XIN, B, 256, 320, 512, 256);
  resize_linear<<<EWG((long)B * 512 * 1280), 0, stream>>>(XIN, U, B, 512, 16, 20, 32, 40);

  // ---- Stage 3: dx3; xin3 = [dx3, x3, pac(xin3)]; u3 = up2(xin3)
  { REPACK(2, 256, 512);
    conv3x3_wmma<<<dim3(B * 1280 / 64, 256 / 64), 256, 0, stream>>>(U, WH, WL, Bc[2], T, B, 512, 32, 40, 256); }
  gn_relu<<<dim3(B, 16), 1024, 0, stream>>>(T, Ga[2], Be[2], XIN, 256, 16, 1280, 1024);
  concat_copy<<<EWG((long)B * 256 * 1280), 0, stream>>>(x3, XIN, B, 256, 1280, 1024, 256);
  resize_linear<<<EWG((long)B * 4 * 1280), 0, stream>>>(emb, Gd, B, 4, 256, 320, 32, 40);
  guide_mask<<<EWG((long)B * 4 * 1280), 0, stream>>>(Gd, mask, B, 32, 40, 256, 320);
  {
    const int dils[3] = {5, 2, 1};
    for (int c = 0; c < 3; ++c)
      pac_pool<<<EWG((long)B * 1280), 0, stream>>>(XIN, Gd, XIN, B, 1024, 512, 32, 40,
                                                   dils[c], 1.f / 3.f, c == 0);
  }
  resize_linear<<<EWG((long)B * 1024 * 5120), 0, stream>>>(XIN, U, B, 1024, 32, 40, 64, 80);

  // ---- Stage 4: dx4; xin4 = [dx4, x2, pac(xin4)]; u4 = up2(xin4)
  { REPACK(3, 128, 1024);
    conv3x3_wmma<<<dim3(B * 5120 / 64, 128 / 64), 256, 0, stream>>>(U, WH, WL, Bc[3], T, B, 1024, 64, 80, 128); }
  gn_relu<<<dim3(B, 8), 1024, 0, stream>>>(T, Ga[3], Be[3], XIN, 128, 8, 5120, 512);
  concat_copy<<<EWG((long)B * 128 * 5120), 0, stream>>>(x2, XIN, B, 128, 5120, 512, 128);
  resize_linear<<<EWG((long)B * 4 * 5120), 0, stream>>>(emb, Gd, B, 4, 256, 320, 64, 80);
  guide_mask<<<EWG((long)B * 4 * 5120), 0, stream>>>(Gd, mask, B, 64, 80, 256, 320);
  {
    const int dils[3] = {10, 5, 2};
    for (int c = 0; c < 3; ++c)
      pac_pool<<<EWG((long)B * 5120), 0, stream>>>(XIN, Gd, XIN, B, 512, 256, 64, 80,
                                                   dils[c], 1.f / 3.f, c == 0);
  }
  resize_linear<<<EWG((long)B * 512 * 20480), 0, stream>>>(XIN, U, B, 512, 64, 80, 128, 160);

  // ---- Stage 5: dx5; xin5 = [dx5, x1, pac(xin5)]; u5 = up2(xin5)
  { REPACK(4, 64, 512);
    conv3x3_wmma<<<dim3(B * 20480 / 64, 64 / 64), 256, 0, stream>>>(U, WH, WL, Bc[4], T, B, 512, 128, 160, 64); }
  gn_relu<<<dim3(B, 4), 1024, 0, stream>>>(T, Ga[4], Be[4], XIN, 64, 4, 20480, 256);
  concat_copy<<<EWG((long)B * 64 * 20480), 0, stream>>>(x1, XIN, B, 64, 20480, 256, 64);
  resize_linear<<<EWG((long)B * 4 * 20480), 0, stream>>>(emb, Gd, B, 4, 256, 320, 128, 160);
  guide_mask<<<EWG((long)B * 4 * 20480), 0, stream>>>(Gd, mask, B, 128, 160, 256, 320);
  {
    const int dils[3] = {20, 10, 5};
    for (int c = 0; c < 3; ++c)
      pac_pool<<<EWG((long)B * 20480), 0, stream>>>(XIN, Gd, XIN, B, 256, 128, 128, 160,
                                                    dils[c], 1.f / 3.f, c == 0);
  }
  resize_linear<<<EWG((long)B * 256 * 81920), 0, stream>>>(XIN, U, B, 256, 128, 160, 256, 320);

  // ---- Stage 6: dx6 = relu(gn4(conv6(u5))); out = conv_f(edge_pad(dx6))
  { REPACK(5, 64, 256);
    conv3x3_wmma<<<dim3(B * 81920 / 64, 64 / 64), 256, 0, stream>>>(U, WH, WL, Bc[5], T, B, 256, 256, 320, 64); }
  gn_relu<<<dim3(B, 4), 1024, 0, stream>>>(T, Ga[5], Be[5], XIN, 64, 4, 81920, 64);
  final_conv<<<EWG((long)B * 81920), 0, stream>>>(XIN, wf, bf, out, B, 256, 320);
#undef REPACK
#undef EWG
}
